// Cross_MultiAttention_1297080123811
// MI455X (gfx1250) — matrix operand order repaired and checked by reading, not compiled, not run on hardware
//
#include <hip/hip_runtime.h>

// ---------------------------------------------------------------------------
// CDNA5 (gfx1250) fused cross-attention with sparsemax.
//   Stage A: Q/K/V projections, f16 outputs. Async (ASYNCcnt) double-buffered
//            f32 LDS staging -> f16 fragments -> v_wmma_f32_16x16x32_f16.
//            Q,K stored [b,h,s,d]; V stored transposed [b,h,d,sk].
//   Stage B: per 16-row query slab: S = QK^T/32, sparsemax via bisection
//            (no sort), att -> HBM once (non-temporal), O = att @ V via
//            LDS-staged f16 att tile. Deterministic reductions only.
// ---------------------------------------------------------------------------

typedef __attribute__((ext_vector_type(16))) _Float16     v16h;
typedef __attribute__((ext_vector_type(8)))  float        v8f;
typedef __attribute__((ext_vector_type(4)))  float        v4f;
typedef __attribute__((ext_vector_type(4)))  unsigned int v4u;

union U16h { v16h h; v4u q[2]; };

#define E_DIM 1024
#define H_DIM 16
#define D_DIM 64
#define B_DIM 8
#define S_DIM 1024
#define SCALE 0.03125f   // E^-0.5

#ifndef __has_builtin
#define __has_builtin(x) 0
#endif

#if __has_builtin(__builtin_amdgcn_global_load_async_to_lds_b128)
#define ASYNC_COPY 1
typedef __attribute__((ext_vector_type(4))) int v4i;
typedef __attribute__((address_space(1))) v4i* gv4i_p;   // global src
typedef __attribute__((address_space(3))) v4i* lv4i_p;   // LDS dst
#endif

__device__ __forceinline__ void cp16_async(const float* g, float* l) {
#ifdef ASYNC_COPY
  __builtin_amdgcn_global_load_async_to_lds_b128(
      (gv4i_p)(v4i*)g, (lv4i_p)(v4i*)l, 0, 0);
#else
  *(v4f*)l = *(const v4f*)g;
#endif
}

__device__ __forceinline__ void stage_wait_async() {
#ifdef ASYNC_COPY
#if __has_builtin(__builtin_amdgcn_s_wait_asynccnt)
  __builtin_amdgcn_s_wait_asynccnt(0);
#else
  asm volatile("s_wait_asynccnt 0x0" ::: "memory");
#endif
#endif
}

// Build a v16h fragment from two runs of 8 consecutive f32 in LDS.
__device__ __forceinline__ v16h frag_from_f32(const float* p0, const float* p1) {
  v4f x0 = *(const v4f*)p0;
  v4f x1 = *(const v4f*)(p0 + 4);
  v4f x2 = *(const v4f*)p1;
  v4f x3 = *(const v4f*)(p1 + 4);
  v16h h;
  for (int i = 0; i < 4; ++i) {
    h[i]      = (_Float16)x0[i];
    h[4 + i]  = (_Float16)x1[i];
    h[8 + i]  = (_Float16)x2[i];
    h[12 + i] = (_Float16)x3[i];
  }
  return h;
}

// ---------------------------------------------------------------------------
// Kernel A: y[s][o] = sum_e x[s][e]*W[o][e] + b[o]
// 256 thr (8 waves), tile 128x128, K-step 32, double-buffered async staging.
// Wave grid 2x4: each wave 64x32 out = 4x2 C-frags.
// LDS tiles are f32, row stride 36 (padded: gcd(36,64)=4 -> conflict-free).
// ---------------------------------------------------------------------------
#define ASTRIDE 36
#define TILE_F  (128 * ASTRIDE)

__global__ __launch_bounds__(256) void qkv_proj_kernel(
    const float* __restrict__ ctx, const float* __restrict__ img,
    const float* __restrict__ Wq, const float* __restrict__ bq,
    const float* __restrict__ Wk, const float* __restrict__ bk,
    const float* __restrict__ Wv, const float* __restrict__ bv,
    _Float16* __restrict__ Qo, _Float16* __restrict__ Ko,
    _Float16* __restrict__ Vto)
{
  const int which = blockIdx.z;                    // 0=Q,1=K,2=V
  const float* x    = (which == 0) ? ctx : img;
  const float* W    = (which == 0) ? Wq : (which == 1 ? Wk : Wv);
  const float* bias = (which == 0) ? bq : (which == 1 ? bk : bv);

  const int tid  = threadIdx.x;
  const int wave = tid >> 5;
  const int lane = tid & 31;
  const int ln   = lane & 15;
  const int mb   = (lane >> 4) * 8;
  const int wm   = wave >> 2;                      // 0..1
  const int wn   = wave & 3;                       // 0..3
  const int m0   = blockIdx.x * 128;               // rows over B*S = 8192
  const int n0   = blockIdx.y * 128;               // cols over E = 1024

  __shared__ __align__(16) float As[2 * TILE_F];   // 2 x 128 x 36 f32
  __shared__ __align__(16) float Bs[2 * TILE_F];

  const int row   = tid >> 1;                      // 0..127
  const int halfc = tid & 1;                       // which 16-float chunk
  const float* xrow = x + (size_t)(m0 + row) * E_DIM + halfc * 16;
  const float* wrow = W + (size_t)(n0 + row) * E_DIM + halfc * 16;
  const int ldst = row * ASTRIDE + halfc * 16;

  v8f acc[4][2] = {};

  // prologue: issue copies for k-step 0 into buffer 0
  for (int q = 0; q < 4; ++q) {
    cp16_async(xrow + q * 4, &As[ldst + q * 4]);
    cp16_async(wrow + q * 4, &Bs[ldst + q * 4]);
  }

  for (int step = 0; step < 32; ++step) {
    float* Ac = &As[(step & 1) * TILE_F];
    float* Bc = &Bs[(step & 1) * TILE_F];

    stage_wait_async();            // this wave's copies into current buffer done
    __syncthreads();               // everyone's copies visible

    if (step < 31) {               // issue next tile into the other buffer
      float* An = &As[((step + 1) & 1) * TILE_F];
      float* Bn = &Bs[((step + 1) & 1) * TILE_F];
      const int k1 = (step + 1) * 32;
      for (int q = 0; q < 4; ++q) {
        cp16_async(xrow + k1 + q * 4, &An[ldst + q * 4]);
        cp16_async(wrow + k1 + q * 4, &Bn[ldst + q * 4]);
      }
    }

    // B-frags: lane holds col n, halves {0-7,16-23} (lo) / {8-15,24-31} (hi)
    U16h bfr[2];
    {
      const int koff = (lane < 16) ? 0 : 8;
      for (int j = 0; j < 2; ++j) {
        const int bn = wn * 32 + j * 16 + ln;
        const float* p = &Bc[bn * ASTRIDE + koff];
        bfr[j].h = frag_from_f32(p, p + 16);
      }
    }
    // A-frags: lane holds row m, halves {0-7,16-23} (lo) / {8-15,24-31} (hi)
    {
      const int aoff = (lane < 16) ? 0 : 8;
      for (int i = 0; i < 4; ++i) {
        const int am = wm * 64 + i * 16 + ln;
        const float* p = &Ac[am * ASTRIDE + aoff];
        U16h afr;
        afr.h = frag_from_f32(p, p + 16);
        for (int j = 0; j < 2; ++j)
          acc[i][j] = __builtin_amdgcn_wmma_f32_16x16x32_f16(
              false, afr.h, false, bfr[j].h, (short)0, acc[i][j], false, false);
      }
    }
    __syncthreads();               // all reads of current buffer complete
  }

  // epilogue: bias, f16, scatter to layout
  for (int i = 0; i < 4; ++i) {
    const int sgb = m0 + wm * 64 + i * 16 + mb;
    for (int j = 0; j < 2; ++j) {
      const int o  = n0 + wn * 32 + j * 16 + ln;
      const float bo = bias[o];
      const int hh = o >> 6, dd = o & 63;
      for (int r = 0; r < 8; ++r) {
        const int s  = sgb + r;
        const int bb = s >> 10, ss = s & 1023;
        const float val = acc[i][j][r] + bo;
        if (which == 2) {
          Vto[(((size_t)bb * H_DIM + hh) * D_DIM + dd) * S_DIM + ss] = (_Float16)val;
        } else {
          _Float16* dst = (which == 0) ? Qo : Ko;
          dst[(((size_t)bb * H_DIM + hh) * S_DIM + ss) * D_DIM + dd] = (_Float16)val;
        }
      }
    }
  }
}

// ---------------------------------------------------------------------------
// Kernel B: fused scores + sparsemax + att@V for one 16-row query slab.
// 8 waves: wave w owns score cols [w*128, w*128+128) and PV k-slice the same.
// ---------------------------------------------------------------------------
__global__ __launch_bounds__(256) void attn_kernel(
    const _Float16* __restrict__ Q, const _Float16* __restrict__ K,
    const _Float16* __restrict__ Vt,
    float* __restrict__ outp, float* __restrict__ attp)
{
  const int tid  = threadIdx.x;
  const int wave = tid >> 5;
  const int lane = tid & 31;
  const int ln   = lane & 15;
  const int hi16 = lane >> 4;
  const int mb   = hi16 * 8;

  const int sq0 = blockIdx.x * 16;
  const int h   = blockIdx.y;
  const int b   = blockIdx.z;
  const size_t bh = (size_t)b * H_DIM + h;

  const _Float16* Qp = Q  + (bh * S_DIM + sq0) * D_DIM;
  const _Float16* Kp = K  + bh * S_DIM * D_DIM;
  const _Float16* Vp = Vt + bh * D_DIM * S_DIM;
  float* attRow = attp + (bh * S_DIM + sq0) * S_DIM;

  __shared__ __align__(16) unsigned char smem[16 * 1032 * 2 + 1536];
  _Float16 (*attS)[1032] = (_Float16(*)[1032])smem;        // 16 x 1032 f16
  float* red  = (float*)(smem + 16 * 1032 * 2);            // [8][16]
  float* red2 = red + 128;                                 // [8][16]
  float* loA  = red2 + 128;                                // [16]
  float* hiA  = loA + 16;                                  // [16]
  float* tauA = hiA + 16;                                  // [16]

  // ---- scores: S = Q K^T * scale ----
  U16h aq[2];
  for (int kc = 0; kc < 2; ++kc) {
    const _Float16* p = Qp + (size_t)ln * D_DIM + kc * 32 + hi16 * 8;
    aq[kc].q[0] = *(const v4u*)p;
    aq[kc].q[1] = *(const v4u*)(p + 16);
  }
  v8f sf[8] = {};
  const int wn0 = wave * 128;
  for (int t = 0; t < 8; ++t) {
    const int n = wn0 + t * 16 + ln;
    for (int kc = 0; kc < 2; ++kc) {
      const _Float16* p = Kp + (size_t)n * D_DIM + kc * 32 + hi16 * 8;
      U16h bf;
      bf.q[0] = *(const v4u*)p;
      bf.q[1] = *(const v4u*)(p + 16);
      sf[t] = __builtin_amdgcn_wmma_f32_16x16x32_f16(
          false, aq[kc].h, false, bf.h, (short)0, sf[t], false, false);
    }
  }
  for (int t = 0; t < 8; ++t)
    for (int r = 0; r < 8; ++r) sf[t][r] *= SCALE;

  // ---- row max (16-lane half reduce, then cross-wave) ----
  float pm[8];
  for (int r = 0; r < 8; ++r) {
    float m = sf[0][r];
    for (int t = 1; t < 8; ++t) m = fmaxf(m, sf[t][r]);
    pm[r] = m;
  }
  for (int d = 1; d < 16; d <<= 1)
    for (int r = 0; r < 8; ++r) pm[r] = fmaxf(pm[r], __shfl_xor(pm[r], d, 32));
  if ((lane & 15) == 0)
    for (int r = 0; r < 8; ++r) red[wave * 16 + mb + r] = pm[r];
  __syncthreads();
  if (tid < 16) {
    float m = red[tid];
    for (int w = 1; w < 8; ++w) m = fmaxf(m, red[w * 16 + tid]);
    loA[tid] = m - 1.0f;     // sum(relu(z-lo)) >= 1
    hiA[tid] = m;            // sum(relu(z-hi)) == 0
  }
  __syncthreads();

  // ---- bisection on tau: sum(max(z - tau, 0)) = 1 ----
  for (int it = 0; it < 30; ++it) {
    if (tid < 16) tauA[tid] = 0.5f * (loA[tid] + hiA[tid]);
    __syncthreads();
    float ps[8];
    for (int r = 0; r < 8; ++r) {
      const float tau = tauA[mb + r];
      float s = 0.f;
      for (int t = 0; t < 8; ++t) s += fmaxf(sf[t][r] - tau, 0.f);
      ps[r] = s;
    }
    for (int d = 1; d < 16; d <<= 1)
      for (int r = 0; r < 8; ++r) ps[r] += __shfl_xor(ps[r], d, 32);
    if ((lane & 15) == 0)
      for (int r = 0; r < 8; ++r) red[wave * 16 + mb + r] = ps[r];
    __syncthreads();
    if (tid < 16) {
      float s = 0.f;
      for (int w = 0; w < 8; ++w) s += red[w * 16 + tid];
      if (s >= 1.0f) loA[tid] = tauA[tid]; else hiA[tid] = tauA[tid];
    }
    __syncthreads();
  }

  // ---- exact tau on isolated support: tau = (sum_support - 1) / k ----
  if (tid < 16) tauA[tid] = 0.5f * (loA[tid] + hiA[tid]);
  __syncthreads();
  {
    float ps[8], pc[8];
    for (int r = 0; r < 8; ++r) {
      const float thr = tauA[mb + r];
      float s = 0.f, c = 0.f;
      for (int t = 0; t < 8; ++t) {
        const float z = sf[t][r];
        if (z > thr) { s += z; c += 1.f; }
      }
      ps[r] = s; pc[r] = c;
    }
    for (int d = 1; d < 16; d <<= 1)
      for (int r = 0; r < 8; ++r) {
        ps[r] += __shfl_xor(ps[r], d, 32);
        pc[r] += __shfl_xor(pc[r], d, 32);
      }
    if ((lane & 15) == 0)
      for (int r = 0; r < 8; ++r) {
        red[wave * 16 + mb + r]  = ps[r];
        red2[wave * 16 + mb + r] = pc[r];
      }
  }
  __syncthreads();
  if (tid < 16) {
    float s = 0.f, c = 0.f;
    for (int w = 0; w < 8; ++w) { s += red[w * 16 + tid]; c += red2[w * 16 + tid]; }
    tauA[tid] = (s - 1.0f) / fmaxf(c, 1.0f);
  }
  __syncthreads();

  // ---- att = relu(z - tau): NT-store to HBM + f16 LDS tile ----
  {
    float taur[8];
    for (int r = 0; r < 8; ++r) taur[r] = tauA[mb + r];
    for (int t = 0; t < 8; ++t) {
      const int n = wn0 + t * 16 + ln;
      for (int r = 0; r < 8; ++r) {
        const int m = mb + r;
        const float v = fmaxf(sf[t][r] - taur[r], 0.f);
        __builtin_nontemporal_store(v, &attRow[(size_t)m * S_DIM + n]);
        attS[m][n] = (_Float16)v;
      }
    }
  }
  __syncthreads();

  // ---- O += att(16 x k-slice) @ V(k-slice x 64), wave k-slice = 128 ----
  v8f of[4] = {};
  const int wk0 = wave * 128;
  for (int kk = 0; kk < 4; ++kk) {
    const int cb = wk0 + kk * 32;
    U16h af;
    const _Float16* p = &attS[ln][cb + hi16 * 8];
    af.q[0] = *(const v4u*)p;
    af.q[1] = *(const v4u*)(p + 16);
    for (int t2 = 0; t2 < 4; ++t2) {
      const int dcol = t2 * 16 + ln;
      const _Float16* pv = Vp + (size_t)dcol * S_DIM + cb + hi16 * 8;
      U16h bf;
      bf.q[0] = *(const v4u*)pv;
      bf.q[1] = *(const v4u*)(pv + 16);
      of[t2] = __builtin_amdgcn_wmma_f32_16x16x32_f16(
          false, af.h, false, bf.h, (short)0, of[t2], false, false);
    }
  }
  __syncthreads();   // attS reads done -> safe to alias

  float* pout = (float*)smem;  // [8][16][64] aliases attS
  for (int t2 = 0; t2 < 4; ++t2)
    for (int r = 0; r < 8; ++r)
      pout[(wave * 16 + mb + r) * 64 + t2 * 16 + ln] = of[t2][r];
  __syncthreads();

  for (int j = 0; j < 4; ++j) {
    const int idx = tid + j * 256;         // 0..1023
    const int m = idx >> 6, d = idx & 63;
    float s = 0.f;
    for (int w = 0; w < 8; ++w) s += pout[(w * 16 + m) * 64 + d];
    __builtin_nontemporal_store(
        s, &outp[((size_t)b * S_DIM + sq0 + m) * E_DIM + h * D_DIM + d]);
  }
}

// ---------------------------------------------------------------------------
extern "C" void kernel_launch(void* const* d_in, const int* in_sizes, int n_in,
                              void* d_out, int out_size, void* d_ws, size_t ws_size,
                              hipStream_t stream) {
  (void)in_sizes; (void)n_in; (void)out_size; (void)ws_size;

  const float* ctx = (const float*)d_in[0];
  const float* img = (const float*)d_in[1];
  const float* Wq  = (const float*)d_in[2];
  const float* bq  = (const float*)d_in[3];
  const float* Wk  = (const float*)d_in[4];
  const float* bk  = (const float*)d_in[5];
  const float* Wv  = (const float*)d_in[6];
  const float* bv  = (const float*)d_in[7];

  const size_t qkv_elems = (size_t)B_DIM * H_DIM * S_DIM * D_DIM;
  _Float16* Qh  = (_Float16*)d_ws;
  _Float16* Kh  = Qh + qkv_elems;
  _Float16* Vth = Kh + qkv_elems;

  float* outp = (float*)d_out;
  float* attp = outp + (size_t)B_DIM * S_DIM * E_DIM;

  qkv_proj_kernel<<<dim3(64, 8, 3), 256, 0, stream>>>(
      ctx, img, Wq, bq, Wk, bk, Wv, bv, Qh, Kh, Vth);

  attn_kernel<<<dim3(S_DIM / 16, H_DIM, B_DIM), 256, 0, stream>>>(
      Qh, Kh, Vth, outp, attp);
}
